// StepModel_12506944766383
// MI455X (gfx1250) — matrix operand order repaired and checked by reading, not compiled, not run on hardware
//
#include <hip/hip_runtime.h>
#include <math.h>

// ---------------------------------------------------------------------------
// CDNA5 (gfx1250) PointNet++ "StepModel" forward pass.
// GEMM core: v_wmma_f32_16x16x32_bf16, operands pre-packed in global memory
// in exact WMMA fragment layout (bf16). No LDS / no barriers / no branches in
// the GEMM hot loop; L2 (192MB) provides operand reuse. BN + activation are
// fused into the GEMM epilogue. Activations are channel-major: M[c][b*N+n].
// All column counts are multiples of 64 and all channel counts are padded to
// multiples of 32, so the GEMM has zero bounds checks.
// ---------------------------------------------------------------------------

typedef __bf16 bf16;
typedef __attribute__((ext_vector_type(8)))  __bf16 v8bf;
typedef __attribute__((ext_vector_type(16))) __bf16 v16bf;
typedef __attribute__((ext_vector_type(8)))  float  v8f;

// ------------------------- packing kernels --------------------------------

// Pack W (Cout x Cin, f32 row-major) into A-fragment-ordered bf16:
//   Wp[(((mt*kT + kt)*32 + lane)*16 + j]
// A 16x32 bf16 fragment: lane = {half = lane>>4, m = lane&15},
//   element j -> K = kt*32 + half*16 + j, M = mt*16 + m.
__global__ void pack_w(const float* __restrict__ W, bf16* __restrict__ Wp,
                       int Cout, int Cin, int Mpad, int kT)
{
  long g = (long)blockIdx.x * blockDim.x + threadIdx.x;
  long tot = (long)(Mpad >> 4) * kT * 32;
  if (g >= tot) return;
  int lane = (int)(g & 31);
  long fk = g >> 5;                 // mt*kT + kt
  int kt = (int)(fk % kT);
  int mt = (int)(fk / kT);
  int half = lane >> 4, m = lane & 15;
  int co = mt * 16 + m;
  bf16* o = Wp + g * 16;
#pragma unroll
  for (int j = 0; j < 16; ++j) {
    int k = kt * 32 + half * 16 + j;
    float v = (co < Cout && k < Cin) ? W[(long)co * Cin + k] : 0.f;
    o[j] = (bf16)v;
  }
}

// Pack X (Kpad x P f32, pad rows zero) into B-fragment-friendly bf16:
//   Xp[((long)kt*P + p)*32 + kk] = X[(kt*32+kk)*P + p]
// B 32x16 fragment: lane {n=lane&15, half=lane>>4}, element j -> K=half*16+j,
// i.e. 16 contiguous bf16 at offset ((kt*P+p)*32 + half*16).
__global__ void pack_x(const float* __restrict__ X, bf16* __restrict__ Xp,
                       int kT, long P)
{
  long g = (long)blockIdx.x * blockDim.x + threadIdx.x;
  if (g >= (long)kT * P) return;
  long p = g % P;
  int kt = (int)(g / P);
  bf16* o = Xp + ((long)kt * P + p) * 32;
  const float* x = X + (long)kt * 32 * P + p;
#pragma unroll
  for (int k = 0; k < 32; ++k) o[k] = (bf16)x[(long)k * P];
}

// ------------------------- WMMA GEMM + BN + act ----------------------------
// Y (Mpad x P) = act( sc[co] * (W @ X) + sh[co] )
// Block = 128 threads (4 waves). Wave wv owns columns [p0+16*wv, +16) and
// computes TWO 16x16 tiles (M = blockIdx.y*32 .. +32) sharing one B fragment.
__global__ __launch_bounds__(128) void gemm_packed_wmma(
    const bf16* __restrict__ Wp, const bf16* __restrict__ Xp,
    float* __restrict__ Y,
    const float* __restrict__ sc, const float* __restrict__ sh,
    int kT, int P, int act /*0 none,1 relu,2 sigmoid*/)
{
  const int t = threadIdx.x;
  const int lane = t & 31, wv = t >> 5;
  const int half = lane >> 4, ml = lane & 15;
  const int p = blockIdx.x * 64 + wv * 16 + ml;
  const int mt0 = blockIdx.y * 2;

  v8f acc0 = {}, acc1 = {};
  const bf16* a0p = Wp + (((long)(mt0 + 0) * kT) * 32 + lane) * 16;
  const bf16* a1p = Wp + (((long)(mt0 + 1) * kT) * 32 + lane) * 16;
  const bf16* bp  = Xp + (long)p * 32 + half * 16;

  for (int kt = 0; kt < kT; ++kt) {
    v8bf a0l = *(const v8bf*)(a0p);
    v8bf a0h = *(const v8bf*)(a0p + 8);
    v8bf a1l = *(const v8bf*)(a1p);
    v8bf a1h = *(const v8bf*)(a1p + 8);
    v8bf bl  = *(const v8bf*)(bp);
    v8bf bh  = *(const v8bf*)(bp + 8);
    v16bf a0 = __builtin_shufflevector(a0l, a0h, 0,1,2,3,4,5,6,7,8,9,10,11,12,13,14,15);
    v16bf a1 = __builtin_shufflevector(a1l, a1h, 0,1,2,3,4,5,6,7,8,9,10,11,12,13,14,15);
    v16bf bb = __builtin_shufflevector(bl,  bh,  0,1,2,3,4,5,6,7,8,9,10,11,12,13,14,15);
    acc0 = __builtin_amdgcn_wmma_f32_16x16x32_bf16(false, a0, false, bb,
                                                   (short)0, acc0, false, false);
    acc1 = __builtin_amdgcn_wmma_f32_16x16x32_bf16(false, a1, false, bb,
                                                   (short)0, acc1, false, false);
    a0p += 32 * 16;                    // next ktile of A frags
    a1p += 32 * 16;
    bp  += (long)P * 32;               // next ktile of B frags
  }

  // D layout: VGPR v -> M = v + half*8 ; N = ml
#pragma unroll
  for (int v = 0; v < 8; ++v) {
    int co0 = mt0 * 16 + v + half * 8;
    int co1 = co0 + 16;
    float y0 = acc0[v] * sc[co0] + sh[co0];
    float y1 = acc1[v] * sc[co1] + sh[co1];
    if (act == 1)      { y0 = fmaxf(y0, 0.f); y1 = fmaxf(y1, 0.f); }
    else if (act == 2) { y0 = 1.f / (1.f + __expf(-y0));
                         y1 = 1.f / (1.f + __expf(-y1)); }
    Y[(long)co0 * P + p] = y0;
    Y[(long)co1 * P + p] = y1;
  }
}

// ------------------------- support kernels --------------------------------

// fold eval-BN into per-channel scale/shift (padded channels -> 0)
__global__ void prep_ss(const float* b, const float* g, const float* be,
                        const float* rm, const float* rv, int C, int Mpad,
                        int bn, float* sc, float* sh)
{
  int i = blockIdx.x * blockDim.x + threadIdx.x;
  if (i >= Mpad) return;
  if (i >= C) { sc[i] = 0.f; sh[i] = 0.f; return; }
  if (bn) {
    float inv = g[i] * rsqrtf(rv[i] + 1e-5f);
    sc[i] = inv;
    sh[i] = (b[i] - rm[i]) * inv + be[i];
  } else {
    sc[i] = 1.f;
    sh[i] = b[i];
  }
}

// (B,C,N) batch-major -> channel-major (C, B*N)
__global__ void bcn_to_cm(const float* __restrict__ s, float* __restrict__ d,
                          int B, int C, int N)
{
  long g = (long)blockIdx.x * blockDim.x + threadIdx.x;
  long tot = (long)B * C * N;
  if (g >= tot) return;
  int n = (int)(g % N);
  int c = (int)((g / N) % C);
  int b = (int)(g / ((long)C * N));
  d[(long)c * B * N + (long)b * N + n] = s[g];
}

// farthest point sampling on channel-major xyz (3, B*Np). one block per batch.
__global__ __launch_bounds__(256) void fps_kernel(const float* __restrict__ xyz,
                                                  int B, int Np, int S, int* out)
{
  __shared__ float dist[2048];
  __shared__ float rvv[256];
  __shared__ int   rii[256];
  __shared__ int   s_last;
  const int b = blockIdx.x, t = threadIdx.x;
  const long cols = (long)B * Np;
  for (int i = t; i < Np; i += 256) dist[i] = 1e10f;
  if (t == 0) { s_last = 0; out[b * S] = 0; }
  __syncthreads();
  for (int it = 1; it < S; ++it) {
    const int last = s_last;
    float lx = xyz[0 * cols + (long)b * Np + last];
    float ly = xyz[1 * cols + (long)b * Np + last];
    float lz = xyz[2 * cols + (long)b * Np + last];
    float bestv = -1.f; int besti = 0x7fffffff;
    for (int i = t; i < Np; i += 256) {
      float dx = xyz[0 * cols + (long)b * Np + i] - lx;
      float dy = xyz[1 * cols + (long)b * Np + i] - ly;
      float dz = xyz[2 * cols + (long)b * Np + i] - lz;
      float d2 = dx * dx + dy * dy + dz * dz;
      float nd = fminf(dist[i], d2);
      dist[i] = nd;
      if (nd > bestv) { bestv = nd; besti = i; }
    }
    rvv[t] = bestv; rii[t] = besti;
    __syncthreads();
    for (int off = 128; off; off >>= 1) {
      if (t < off) {
        if (rvv[t + off] > rvv[t] ||
            (rvv[t + off] == rvv[t] && rii[t + off] < rii[t])) {
          rvv[t] = rvv[t + off]; rii[t] = rii[t + off];
        }
      }
      __syncthreads();
    }
    if (t == 0) { s_last = rii[0]; out[b * S + it] = rii[0]; }
    __syncthreads();
  }
}

__global__ void gather_newxyz(const float* __restrict__ xyz, int Np,
                              const int* __restrict__ fidx, int B, int S,
                              float* __restrict__ nx)
{
  long g = (long)blockIdx.x * blockDim.x + threadIdx.x;
  if (g >= (long)B * S) return;
  int b = (int)(g / S);
  int id = fidx[g];
  long cols = (long)B * Np, nc = (long)B * S;
  for (int d = 0; d < 3; ++d)
    nx[d * nc + g] = xyz[d * cols + (long)b * Np + id];
}

__global__ void ball_query_kernel(const float* __restrict__ xyz, int Np,
                                  const float* __restrict__ nx, int S, int B,
                                  float r2, int K, int* __restrict__ idx)
{
  long g = (long)blockIdx.x * blockDim.x + threadIdx.x;
  if (g >= (long)B * S) return;
  int b = (int)(g / S);
  long cols = (long)B * Np, nc = (long)B * S;
  float cx = nx[g], cy = nx[nc + g], cz = nx[2 * nc + g];
  int* o = idx + g * K;
  int cnt = 0;
  for (int j = 0; j < Np && cnt < K; ++j) {
    float dx = xyz[0 * cols + (long)b * Np + j] - cx;
    float dy = xyz[1 * cols + (long)b * Np + j] - cy;
    float dz = xyz[2 * cols + (long)b * Np + j] - cz;
    if (dx * dx + dy * dy + dz * dz < r2) o[cnt++] = j;
  }
  int first = (cnt > 0) ? o[0] : 0;        // pointnet2 padding semantics
  for (int c = cnt; c < K; ++c) o[c] = first;
}

// build grouped GEMM input G (3+C, B*S*K): rows 0-2 centered xyz, rows 3.. features
__global__ void group_kernel(const float* __restrict__ xyz, int Np,
                             const float* __restrict__ nx,
                             const float* __restrict__ pts, int C,
                             const int* __restrict__ idx, int B, int S, int K,
                             float* __restrict__ G)
{
  long g = (long)blockIdx.x * blockDim.x + threadIdx.x;
  long Pf = (long)B * S * K;
  if (g >= Pf) return;
  int bs = (int)(g / K);
  int b  = bs / S;
  int id = idx[g];
  long cols = (long)B * Np, nc = (long)B * S;
  for (int d = 0; d < 3; ++d)
    G[d * Pf + g] = xyz[d * cols + (long)b * Np + id] - nx[d * nc + bs];
  for (int c = 0; c < C; ++c)
    G[(3 + c) * Pf + g] = pts[(long)c * cols + (long)b * Np + id];
}

__global__ void maxpool_k(const float* __restrict__ T, int Cout, int B, int S,
                          int K, float* __restrict__ outcm)
{
  long g = (long)blockIdx.x * blockDim.x + threadIdx.x;
  long tot = (long)Cout * B * S;
  if (g >= tot) return;
  long Pf = (long)B * S * K;
  const float* p = T + (g / ((long)B * S)) * Pf + (g % ((long)B * S)) * K;
  float m = -3.4e38f;
  for (int k = 0; k < K; ++k) m = fmaxf(m, p[k]);
  outcm[g] = m;
}

__global__ void maxpool_n(const float* __restrict__ T, int Cout, int B, int Nn,
                          float* __restrict__ outcm)
{
  long g = (long)blockIdx.x * blockDim.x + threadIdx.x;
  if (g >= (long)Cout * B) return;
  int co = (int)(g / B), b = (int)(g % B);
  const float* p = T + (long)co * B * Nn + (long)b * Nn;
  float m = -3.4e38f;
  for (int n = 0; n < Nn; ++n) m = fmaxf(m, p[n]);
  outcm[(long)co * B + b] = m;
}

__global__ void three_nn_kernel(const float* __restrict__ x1,
                                const float* __restrict__ x2, int B, int N1,
                                int N2, int* __restrict__ idx,
                                float* __restrict__ w)
{
  long g = (long)blockIdx.x * blockDim.x + threadIdx.x;
  if (g >= (long)B * N1) return;
  int b = (int)(g / N1);
  long c1 = (long)B * N1, c2 = (long)B * N2;
  float px = x1[g], py = x1[c1 + g], pz = x1[2 * c1 + g];
  float d0 = 1e30f, d1 = 1e30f, d2 = 1e30f;
  int i0 = 0, i1 = 0, i2 = 0;
  for (int j = 0; j < N2; ++j) {
    float dx = x2[0 * c2 + (long)b * N2 + j] - px;
    float dy = x2[1 * c2 + (long)b * N2 + j] - py;
    float dz = x2[2 * c2 + (long)b * N2 + j] - pz;
    float d = dx * dx + dy * dy + dz * dz;
    if (d < d0)      { d2=d1;i2=i1; d1=d0;i1=i0; d0=d;i0=j; }
    else if (d < d1) { d2=d1;i2=i1; d1=d;i1=j; }
    else if (d < d2) { d2=d;i2=j; }
  }
  if (N2 < 2) { d1 = d0; i1 = i0; }
  if (N2 < 3) { d2 = d1; i2 = i1; }
  float r0 = 1.f / fmaxf(d0, 1e-10f);
  float r1 = 1.f / fmaxf(d1, 1e-10f);
  float r2 = 1.f / fmaxf(d2, 1e-10f);
  float s = r0 + r1 + r2;
  idx[g * 3 + 0] = i0; idx[g * 3 + 1] = i1; idx[g * 3 + 2] = i2;
  w[g * 3 + 0] = r0 / s; w[g * 3 + 1] = r1 / s; w[g * 3 + 2] = r2 / s;
}

// interpolate points2 (C2, B*N2) onto rows [r0, r0+C2) of G (Rtot, B*N1)
__global__ void interp_rows(const float* __restrict__ p2, int C2, int N2,
                            const int* __restrict__ idx3,
                            const float* __restrict__ w3, int B, int N1,
                            float* __restrict__ G, long Pcols, int r0)
{
  long g = (long)blockIdx.x * blockDim.x + threadIdx.x;
  long tot = (long)C2 * B * N1;
  if (g >= tot) return;
  int c = (int)(g / ((long)B * N1));
  long col = g % ((long)B * N1);
  int b = (int)(col / N1);
  const int*   id = idx3 + col * 3;
  const float* w  = w3 + col * 3;
  long base = (long)c * B * N2 + (long)b * N2;
  float v = w[0] * p2[base + id[0]] + w[1] * p2[base + id[1]] +
            w[2] * p2[base + id[2]];
  G[(long)(r0 + c) * Pcols + col] = v;
}

__global__ void copyf(const float* __restrict__ s, float* __restrict__ d, long n)
{
  long i = (long)blockIdx.x * blockDim.x + threadIdx.x;
  if (i < n) d[i] = s[i];
}
__global__ void mulf(const float* __restrict__ a, const float* __restrict__ b,
                     float* __restrict__ d, long n)
{
  long i = (long)blockIdx.x * blockDim.x + threadIdx.x;
  if (i < n) d[i] = a[i] * b[i];
}
__global__ void gru_comb(const float* __restrict__ z, const float* __restrict__ cur,
                         const float* __restrict__ h, float* __restrict__ o, long n)
{
  long i = (long)blockIdx.x * blockDim.x + threadIdx.x;
  if (i < n) o[i] = (1.f - z[i]) * cur[i] + z[i] * h[i];
}
__global__ void zerof(float* __restrict__ p, long n)
{
  for (long i = (long)blockIdx.x * blockDim.x + threadIdx.x; i < n;
       i += (long)gridDim.x * blockDim.x) p[i] = 0.f;
}

// delta = tanh(x)*scale ; out0 = pc + delta ; out1 = delta   (both (B,3,N))
__global__ void final_out(const float* __restrict__ xcm,
                          const float* __restrict__ pc, float* __restrict__ out,
                          int B, int Nn, float scale)
{
  long g = (long)blockIdx.x * blockDim.x + threadIdx.x;
  long tot = (long)3 * B * Nn;
  if (g >= tot) return;
  int c = (int)(g / ((long)B * Nn));
  long col = g % ((long)B * Nn);
  int b = (int)(col / Nn), n = (int)(col % Nn);
  float delta = tanhf(xcm[g]) * scale;
  long o = ((long)b * 3 + c) * Nn + n;
  out[o]       = pc[o] + delta;
  out[tot + o] = delta;
}

// ------------------------------ host side ---------------------------------

extern "C" void kernel_launch(void* const* d_in, const int* in_sizes, int n_in,
                              void* d_out, int out_size, void* d_ws,
                              size_t ws_size, hipStream_t stream)
{
  constexpr int B = 16, N0 = 2048, S1 = 512, K1 = 32, S2 = 128, K2 = 32;
  const long C0 = (long)B * N0;       // 32768 columns
  const long C1 = (long)B * S1;       // 8192
  const long C2c = (long)B * S2;      // 2048
  const long P1 = (long)B * S1 * K1;  // 262144
  const long P2 = (long)B * S2 * K2;  // 65536

  const float* pc_in = (const float*)d_in[0];
  const float* pl0   = (const float*)d_in[1];
  const float* pl1   = (const float*)d_in[2];
  const float* pl2   = (const float*)d_in[3];
  const float* noise = (const float*)d_in[4];

  float* Wsp = (float*)d_ws;
  size_t off = 0;
  auto alloc = [&](size_t n) -> float* {
    float* p = Wsp + off; off += (n + 63) & ~(size_t)63; return p;
  };

  // ---- params (jax tree order: dict keys sorted; per-conv: b,be,g,rm,rv,w) --
  int pi = 5;
  struct Conv {
    const float *w, *b, *g, *be, *rm, *rv;
    int cin, cout, bn, Mpad, Kpad, kT;
    float *sc, *sh; bf16* wp;
  };
  Conv convs[28]; int nc = 0;
  auto take = [&](int cin, int cout, int bn) -> int {
    Conv c{};
    c.cin = cin; c.cout = cout; c.bn = bn;
    c.Mpad = (cout + 31) & ~31;
    c.Kpad = (cin + 31) & ~31;
    c.kT = c.Kpad >> 5;
    if (bn) {
      c.b  = (const float*)d_in[pi + 0]; c.be = (const float*)d_in[pi + 1];
      c.g  = (const float*)d_in[pi + 2]; c.rm = (const float*)d_in[pi + 3];
      c.rv = (const float*)d_in[pi + 4]; c.w  = (const float*)d_in[pi + 5];
      pi += 6;
    } else {
      c.b = (const float*)d_in[pi + 0]; c.w = (const float*)d_in[pi + 1];
      c.g = c.be = c.rm = c.rv = nullptr; pi += 2;
    }
    c.sc = alloc(c.Mpad); c.sh = alloc(c.Mpad);
    c.wp = (bf16*)alloc(((size_t)c.Mpad * c.Kpad + 1) / 2);
    convs[nc] = c; return nc++;
  };
  // alphabetical: fp1, fp2, fp3, mlp, sa1, sa2, sa3, u1(h,r,z), u2, u3
  int fp1_0 = take(134, 128, 1), fp1_1 = take(128, 128, 1), fp1_2 = take(128, 128, 1);
  int fp2_0 = take(384, 256, 1), fp2_1 = take(256, 128, 1);
  int fp3_0 = take(1280, 256, 1), fp3_1 = take(256, 256, 1);
  int m0 = take(160, 128, 1), m1 = take(128, 64, 1), m2 = take(64, 3, 0);
  int s1_0 = take(6, 64, 1), s1_1 = take(64, 64, 1), s1_2 = take(64, 128, 1);
  int s2_0 = take(131, 128, 1), s2_1 = take(128, 128, 1), s2_2 = take(128, 256, 1);
  int s3_0 = take(259, 256, 1), s3_1 = take(256, 512, 1), s3_2 = take(512, 1024, 1);
  int u1h = take(256, 128, 1), u1r = take(256, 128, 1), u1z = take(256, 128, 1);
  int u2h = take(256, 128, 1), u2r = take(256, 128, 1), u2z = take(256, 128, 1);
  int u3h = take(512, 256, 1), u3r = take(512, 256, 1), u3z = take(512, 256, 1);
  (void)in_sizes; (void)n_in; (void)out_size; (void)ws_size;

  // ---- workspace tensors (channel-major; G matrices have Kpad rows) ----
  float* pc_cm    = alloc(3 * C0);
  float* pl0_cm   = alloc(128 * C0);
  float* pl1_cm   = alloc(128 * C1);
  float* pl2_cm   = alloc(256 * C2c);
  float* noise_cm = alloc(32 * C0);
  int*   fidx1 = (int*)alloc(B * S1);
  float* nx1   = alloc(3 * C1);
  int*   idx1  = (int*)alloc(P1);
  float* G1    = alloc(32 * P1);          // Kpad(6)=32
  float* bufA  = alloc(128 * P1);
  float* bufB  = alloc(128 * P1);
  float* l1p   = alloc(128 * C1);
  int*   fidx2 = (int*)alloc(B * S2);
  float* nx2   = alloc(3 * C2c);
  int*   idx2  = (int*)alloc(P2);
  float* G2    = alloc(160 * P2);         // Kpad(131)=160
  float* l2p   = alloc(256 * C2c);
  float* G3    = alloc(288 * C2c);        // Kpad(259)=288
  float* l3p   = alloc(1024 * B);
  float* l3xyz = alloc(3 * B);
  int*   nnidx = (int*)alloc(3 * C0);
  float* nnw   = alloc(3 * C0);
  float* Gbig  = alloc(160 * C0);         // max concat matrix (Kpad rows)
  float* l2f   = alloc(256 * C2c);
  float* cat   = alloc(256 * C0);         // max GRU concat (u1: 256 x 32768)
  float* zb    = alloc(128 * C0);
  float* rb    = alloc(128 * C0);
  float* hb    = alloc(128 * C0);
  float* l2u   = alloc(256 * C2c);
  float* l1f   = alloc(128 * C1);
  float* l1u   = alloc(128 * C1);
  float* l0f   = alloc(128 * C0);
  float* l0u   = alloc(128 * C0);
  bf16*  Xp    = (bf16*)alloc((size_t)(64 * P1) / 2);  // max Kpad*P (bf16)

  auto gl = [](long n) { return dim3((unsigned)((n + 255) / 256)); };
  auto run_conv = [&](int ci, const float* X, float* Y, long P, int act) {
    const Conv& c = convs[ci];
    pack_x<<<gl((long)c.kT * P), 256, 0, stream>>>(X, Xp, c.kT, P);
    dim3 grid((unsigned)(P / 64), (unsigned)(c.Mpad / 32));
    gemm_packed_wmma<<<grid, 128, 0, stream>>>(c.wp, Xp, Y, c.sc, c.sh,
                                               c.kT, (int)P, act);
  };

  // BN folding + weight packing
  for (int i = 0; i < nc; ++i) {
    const Conv& c = convs[i];
    prep_ss<<<(c.Mpad + 63) / 64, 64, 0, stream>>>(c.b, c.g, c.be, c.rm, c.rv,
                                                   c.cout, c.Mpad, c.bn, c.sc, c.sh);
    pack_w<<<gl((long)(c.Mpad >> 4) * c.kT * 32), 256, 0, stream>>>(
        c.w, c.wp, c.cout, c.cin, c.Mpad, c.kT);
  }

  // layout conversion to channel-major
  bcn_to_cm<<<gl(3 * C0), 256, 0, stream>>>(pc_in, pc_cm, B, 3, N0);
  bcn_to_cm<<<gl(128 * C0), 256, 0, stream>>>(pl0, pl0_cm, B, 128, N0);
  bcn_to_cm<<<gl(128 * C1), 256, 0, stream>>>(pl1, pl1_cm, B, 128, S1);
  bcn_to_cm<<<gl(256 * C2c), 256, 0, stream>>>(pl2, pl2_cm, B, 256, S2);
  bcn_to_cm<<<gl(32 * C0), 256, 0, stream>>>(noise, noise_cm, B, 32, N0);

  // ---- SA1 ----
  fps_kernel<<<B, 256, 0, stream>>>(pc_cm, B, N0, S1, fidx1);
  gather_newxyz<<<gl(C1), 256, 0, stream>>>(pc_cm, N0, fidx1, B, S1, nx1);
  ball_query_kernel<<<gl(C1), 256, 0, stream>>>(pc_cm, N0, nx1, S1, B, 0.04f, K1, idx1);
  group_kernel<<<gl(P1), 256, 0, stream>>>(pc_cm, N0, nx1, pc_cm, 3, idx1, B, S1, K1, G1);
  zerof<<<gl(26 * P1), 256, 0, stream>>>(G1 + 6 * P1, 26 * P1);     // pad rows
  run_conv(s1_0, G1, bufA, P1, 1);
  run_conv(s1_1, bufA, bufB, P1, 1);
  run_conv(s1_2, bufB, bufA, P1, 1);
  maxpool_k<<<gl(128 * C1), 256, 0, stream>>>(bufA, 128, B, S1, K1, l1p);

  // ---- SA2 ----
  fps_kernel<<<B, 256, 0, stream>>>(nx1, B, S1, S2, fidx2);
  gather_newxyz<<<gl(C2c), 256, 0, stream>>>(nx1, S1, fidx2, B, S2, nx2);
  ball_query_kernel<<<gl(C2c), 256, 0, stream>>>(nx1, S1, nx2, S2, B, 0.16f, K2, idx2);
  group_kernel<<<gl(P2), 256, 0, stream>>>(nx1, S1, nx2, l1p, 128, idx2, B, S2, K2, G2);
  zerof<<<gl(29 * P2), 256, 0, stream>>>(G2 + 131 * P2, 29 * P2);   // pad rows
  run_conv(s2_0, G2, bufA, P2, 1);
  run_conv(s2_1, bufA, bufB, P2, 1);
  run_conv(s2_2, bufB, bufA, P2, 1);
  maxpool_k<<<gl(256 * C2c), 256, 0, stream>>>(bufA, 256, B, S2, K2, l2p);

  // ---- SA3 (group-all) ----
  copyf<<<gl(3 * C2c), 256, 0, stream>>>(nx2, G3, 3 * C2c);
  copyf<<<gl(256 * C2c), 256, 0, stream>>>(l2p, G3 + 3 * C2c, 256 * C2c);
  zerof<<<gl(29 * C2c), 256, 0, stream>>>(G3 + 259 * C2c, 29 * C2c); // pad rows
  run_conv(s3_0, G3, bufA, C2c, 1);
  run_conv(s3_1, bufA, bufB, C2c, 1);
  run_conv(s3_2, bufB, bufA, C2c, 1);
  maxpool_n<<<gl(1024 * B), 256, 0, stream>>>(bufA, 1024, B, S2, l3p);
  zerof<<<1, 64, 0, stream>>>(l3xyz, 3 * B);

  // ---- FP3 ----
  three_nn_kernel<<<gl(C2c), 256, 0, stream>>>(nx2, l3xyz, B, S2, 1, nnidx, nnw);
  interp_rows<<<gl(1024 * C2c), 256, 0, stream>>>(l3p, 1024, 1, nnidx, nnw, B, S2, Gbig, C2c, 0);
  copyf<<<gl(256 * C2c), 256, 0, stream>>>(l2p, Gbig + 1024 * C2c, 256 * C2c);
  run_conv(fp3_0, Gbig, bufA, C2c, 1);
  run_conv(fp3_1, bufA, l2f, C2c, 1);

  // ---- unit u3 (C=256) ----
  copyf<<<gl(256 * C2c), 256, 0, stream>>>(l2f, cat, 256 * C2c);
  copyf<<<gl(256 * C2c), 256, 0, stream>>>(pl2_cm, cat + 256 * C2c, 256 * C2c);
  run_conv(u3z, cat, zb, C2c, 2);
  run_conv(u3r, cat, rb, C2c, 2);
  mulf<<<gl(256 * C2c), 256, 0, stream>>>(rb, pl2_cm, cat + 256 * C2c, 256 * C2c);
  run_conv(u3h, cat, hb, C2c, 1);
  gru_comb<<<gl(256 * C2c), 256, 0, stream>>>(zb, l2f, hb, l2u, 256 * C2c);

  // ---- FP2 ----
  three_nn_kernel<<<gl(C1), 256, 0, stream>>>(nx1, nx2, B, S1, S2, nnidx, nnw);
  interp_rows<<<gl(256 * C1), 256, 0, stream>>>(l2u, 256, S2, nnidx, nnw, B, S1, Gbig, C1, 0);
  copyf<<<gl(128 * C1), 256, 0, stream>>>(l1p, Gbig + 256 * C1, 128 * C1);
  run_conv(fp2_0, Gbig, bufA, C1, 1);
  run_conv(fp2_1, bufA, l1f, C1, 1);

  // ---- unit u2 (C=128) ----
  copyf<<<gl(128 * C1), 256, 0, stream>>>(l1f, cat, 128 * C1);
  copyf<<<gl(128 * C1), 256, 0, stream>>>(pl1_cm, cat + 128 * C1, 128 * C1);
  run_conv(u2z, cat, zb, C1, 2);
  run_conv(u2r, cat, rb, C1, 2);
  mulf<<<gl(128 * C1), 256, 0, stream>>>(rb, pl1_cm, cat + 128 * C1, 128 * C1);
  run_conv(u2h, cat, hb, C1, 1);
  gru_comb<<<gl(128 * C1), 256, 0, stream>>>(zb, l1f, hb, l1u, 128 * C1);

  // ---- FP1 (points1 = [xyz; point_cloud]) ----
  three_nn_kernel<<<gl(C0), 256, 0, stream>>>(pc_cm, nx1, B, N0, S1, nnidx, nnw);
  interp_rows<<<gl(128 * C0), 256, 0, stream>>>(l1u, 128, S1, nnidx, nnw, B, N0, Gbig, C0, 0);
  copyf<<<gl(3 * C0), 256, 0, stream>>>(pc_cm, Gbig + 128 * C0, 3 * C0);
  copyf<<<gl(3 * C0), 256, 0, stream>>>(pc_cm, Gbig + 131 * C0, 3 * C0);
  zerof<<<gl(26 * C0), 256, 0, stream>>>(Gbig + 134 * C0, 26 * C0); // pad rows
  run_conv(fp1_0, Gbig, bufA, C0, 1);
  run_conv(fp1_1, bufA, bufB, C0, 1);
  run_conv(fp1_2, bufB, l0f, C0, 1);

  // ---- unit u1 (C=128) ----
  copyf<<<gl(128 * C0), 256, 0, stream>>>(l0f, cat, 128 * C0);
  copyf<<<gl(128 * C0), 256, 0, stream>>>(pl0_cm, cat + 128 * C0, 128 * C0);
  run_conv(u1z, cat, zb, C0, 2);
  run_conv(u1r, cat, rb, C0, 2);
  mulf<<<gl(128 * C0), 256, 0, stream>>>(rb, pl0_cm, cat + 128 * C0, 128 * C0);
  run_conv(u1h, cat, hb, C0, 1);
  gru_comb<<<gl(128 * C0), 256, 0, stream>>>(zb, l0f, hb, l0u, 128 * C0);

  // ---- head MLP + tanh ----
  copyf<<<gl(128 * C0), 256, 0, stream>>>(l0u, Gbig, 128 * C0);
  copyf<<<gl(32 * C0), 256, 0, stream>>>(noise_cm, Gbig + 128 * C0, 32 * C0);
  run_conv(m0, Gbig, bufA, C0, 1);
  run_conv(m1, bufA, bufB, C0, 1);
  run_conv(m2, bufB, bufA, C0, 0);
  final_out<<<gl(3 * C0), 256, 0, stream>>>(bufA, pc_in, (float*)d_out, B, N0, 0.1f);
}
